// CNNAttention_17334488007340
// MI455X (gfx1250) — matrix operand order repaired and checked by reading, not compiled, not run on hardware
//
#include <hip/hip_runtime.h>

// ---------------------------------------------------------------------------
// CDNA5 (gfx1250) wave32 WMMA implementation of cosine-attention CNN block.
// Heavy math: v_wmma_f32_16x16x32_bf16. V tiles staged with
// global_load_async_to_lds_b128 (ASYNCcnt path). Native bf16 converts.
// ---------------------------------------------------------------------------

typedef __attribute__((ext_vector_type(16))) __bf16 v16bf;
typedef __attribute__((ext_vector_type(8)))  float  v8f;

union FragBF {            // 16 bf16 = 8 VGPRs (A or B operand)
    v16bf v;
    uint4 q[2];
    unsigned int w[8];
    unsigned short h[16];
};
union AccF {              // 16x16 f32 C/D accumulator = 8 VGPRs
    v8f v;
    float f[8];
};

__device__ __forceinline__ unsigned short f2bf(float f) {
    // native fp32 -> bf16 convert (v_cvt_*bf16*), RTNE
    __bf16 h = (__bf16)f;
    return __builtin_bit_cast(unsigned short, h);
}
__device__ __forceinline__ float bf2f(unsigned short h) {
    return __uint_as_float(((unsigned)h) << 16);
}

__device__ __forceinline__ void async_wait0() {
#if __has_builtin(__builtin_amdgcn_s_wait_asynccnt)
    __builtin_amdgcn_s_wait_asynccnt(0);
#else
    asm volatile("s_wait_asynccnt 0" ::: "memory");
#endif
}

// per-lane async copy of 16 bytes global -> LDS (no VGPR round trip)
__device__ __forceinline__ void async_copy_b128(void* lds_dst, const void* gsrc) {
    unsigned ldsoff = (unsigned)(unsigned long long)lds_dst;   // low 32 bits = LDS offset
    asm volatile("global_load_async_to_lds_b128 %0, %1, off"
                 :: "v"(ldsoff), "v"((unsigned long long)gsrc)
                 : "memory");
}

#define NPOS   4096   // H*W
#define CIN    256
#define INNER  512
#define HEADS  8
#define DHEAD  64
#define SMOOTH 1e-4f
#define BN_EPS 1e-5f

// ---------------------------------------------------------------------------
// Kernel 1: qkv[o,p] = sum_c w_qkv[o,c] * x[c,p]    (M=1536, K=256, N=4096)
// Output written straight into bf16 q/k/v buffers, layout [head][pos][d].
// WG = 256 threads = 8 waves; tile 128(M) x 64(N); K-step 32.
// ---------------------------------------------------------------------------
__global__ __launch_bounds__(256) void k_qkv_gemm(
    const float* __restrict__ x,          // [256][4096]
    const float* __restrict__ w,          // [1536][256]
    unsigned short* __restrict__ qb,      // [8][4096][64] bf16
    unsigned short* __restrict__ kb,
    unsigned short* __restrict__ vb)
{
    __shared__ __align__(16) unsigned short at[128][32]; // A tile (o x c) bf16
    __shared__ __align__(16) unsigned short bt[32][64];  // B tile (c x p) bf16

    const int tid  = threadIdx.x;
    const int lane = tid & 31;
    const int wv   = tid >> 5;
    const int mblk = blockIdx.x % 12;
    const int nblk = blockIdx.x / 12;
    const int o0   = mblk * 128;
    const int p0   = nblk * 64;

    AccF acc[4];
#pragma unroll
    for (int i = 0; i < 4; ++i)
#pragma unroll
        for (int r = 0; r < 8; ++r) acc[i].f[r] = 0.0f;

    for (int k0 = 0; k0 < CIN; k0 += 32) {
        __syncthreads();
        {   // stage A: 128x32, 16 elems/thread
            int row = tid >> 1;
            int cb  = (tid & 1) * 16;
            const float* src = w + (o0 + row) * CIN + k0 + cb;
#pragma unroll
            for (int i = 0; i < 16; ++i) at[row][cb + i] = f2bf(src[i]);
        }
        {   // stage B: 32x64, 8 elems/thread
            int row = tid >> 3;
            int cb  = (tid & 7) * 8;
            const float* src = x + (k0 + row) * NPOS + p0 + cb;
#pragma unroll
            for (int i = 0; i < 8; ++i) bt[row][cb + i] = f2bf(src[i]);
        }
        __syncthreads();

        FragBF a;       // A: lane&15 = M row; lane>>4 selects K-halves
        {
            int r   = wv * 16 + (lane & 15);
            int kbo = (lane >> 4) * 8;
            a.q[0] = *(const uint4*)&at[r][kbo];
            a.q[1] = *(const uint4*)&at[r][kbo + 16];
        }
#pragma unroll
        for (int ns = 0; ns < 4; ++ns) {
            FragBF b;   // B: lane&15 = N column; elements = K (table order)
#pragma unroll
            for (int i = 0; i < 16; ++i)
                b.h[i] = bt[(lane >> 4) * 8 + (i & 7) + ((i >> 3) << 4)][ns * 16 + (lane & 15)];
            acc[ns].v = __builtin_amdgcn_wmma_f32_16x16x32_bf16(
                false, a.v, false, b.v, (short)0, acc[ns].v, false, false);
        }
    }

    // Store: lane's 8 accumulator rows are 8 consecutive output channels
    // => 8 consecutive d values inside one head => one b128 bf16 store.
    {
        int rbase = wv * 16 + ((lane >> 4) * 8);
        int oc0   = o0 + rbase;
        int which = oc0 >> 9;           // 0=q 1=k 2=v
        int inner = oc0 & 511;
        int g     = inner >> 6;
        int d0    = inner & 63;         // multiple of 8
        unsigned short* dst = (which == 0) ? qb : ((which == 1) ? kb : vb);
#pragma unroll
        for (int ns = 0; ns < 4; ++ns) {
            int p = p0 + ns * 16 + (lane & 15);
            union { uint4 u; unsigned short h[8]; } pk;
#pragma unroll
            for (int r = 0; r < 8; ++r) pk.h[r] = f2bf(acc[ns].f[r]);
            *(uint4*)&dst[((g * NPOS) + p) * DHEAD + d0] = pk.u;
        }
    }
}

// ---------------------------------------------------------------------------
// Kernel 2: per-position L2 norms of q and k:  qn/kn[g*4096+p]
// ---------------------------------------------------------------------------
__global__ __launch_bounds__(256) void k_norms(
    const unsigned short* __restrict__ qb,
    const unsigned short* __restrict__ kb,
    float* __restrict__ qn, float* __restrict__ kn)
{
    int i = blockIdx.x * 256 + threadIdx.x;         // (g,p) flat, 32768 total
    const unsigned short* qrow = qb + i * DHEAD;
    const unsigned short* krow = kb + i * DHEAD;
    float sq = 0.0f, sk = 0.0f;
#pragma unroll
    for (int d = 0; d < DHEAD; d += 8) {
        uint4 a = *(const uint4*)(qrow + d);
        uint4 b = *(const uint4*)(krow + d);
        const unsigned* aw = (const unsigned*)&a;
        const unsigned* bw = (const unsigned*)&b;
#pragma unroll
        for (int j = 0; j < 4; ++j) {
            float a0 = __uint_as_float(aw[j] << 16);
            float a1 = __uint_as_float(aw[j] & 0xFFFF0000u);
            float b0 = __uint_as_float(bw[j] << 16);
            float b1 = __uint_as_float(bw[j] & 0xFFFF0000u);
            sq += a0 * a0 + a1 * a1;
            sk += b0 * b0 + b1 * b1;
        }
    }
    qn[i] = sqrtf(sq + SMOOTH);
    kn[i] = sqrtf(sk + SMOOTH);
}

// ---------------------------------------------------------------------------
// Kernel 3: streaming cosine attention (no softmax => no global row state).
// Grid: 8 heads x 32 n-blocks of 128 rows. WG = 8 waves; each wave owns a
// 16(n) x 64(d) output tile (4 f32 accumulators). Loop over m in tiles of 32:
//   S(16x32)  = Q(16x64) . K^T            (4 wmma)
//   P         = S / (|q||k| + eps)        (VALU, restaged via per-wave LDS)
//   O(16x64) += P(16x32) . V(32x64)       (4 wmma)
// V tile staged via global_load_async_to_lds_b128; K tile transposed in
// registers while the async copy is in flight.
// O written channel-major [512][4096] so k_proj's B matrix is contiguous.
// ---------------------------------------------------------------------------
__global__ __launch_bounds__(256) void k_attn(
    const unsigned short* __restrict__ qb,
    const unsigned short* __restrict__ kbm,
    const unsigned short* __restrict__ vbm,
    const float* __restrict__ qn,
    const float* __restrict__ kn,
    unsigned short* __restrict__ ob)          // [512][4096] bf16, row = g*64+d
{
    __shared__ __align__(16) unsigned short ktT[64][32];      // K^T tile [d][m]
    __shared__ __align__(16) unsigned short vt[32][64];       // V tile   [m][d]
    __shared__ __align__(16) unsigned short pst[8][16][32];   // per-wave P stage

    const int tid  = threadIdx.x;
    const int lane = tid & 31;
    const int wv   = tid >> 5;
    const int g    = blockIdx.x >> 5;
    const int nblk = blockIdx.x & 31;
    const int n0w  = nblk * 128 + wv * 16;    // wave's first output row

    // Q fragments for the wave's 16 rows: two K=32 chunks of d
    FragBF qa[2];
    {
        int n   = n0w + (lane & 15);
        int kbo = (lane >> 4) * 8;
        const unsigned short* qrow = qb + (g * NPOS + n) * DHEAD;
        qa[0].q[0] = *(const uint4*)(qrow + kbo);
        qa[0].q[1] = *(const uint4*)(qrow + kbo + 16);
        qa[1].q[0] = *(const uint4*)(qrow + 32 + kbo);
        qa[1].q[1] = *(const uint4*)(qrow + 32 + kbo + 16);
    }
    // qn for the 8 rows this lane's accumulator registers cover
    float qn_r[8];
    {
        int nb = n0w + ((lane >> 4) * 8);
#pragma unroll
        for (int r = 0; r < 8; ++r) qn_r[r] = qn[g * NPOS + nb + r];
    }

    AccF oacc[4];
#pragma unroll
    for (int i = 0; i < 4; ++i)
#pragma unroll
        for (int r = 0; r < 8; ++r) oacc[i].f[r] = 0.0f;

    for (int m0 = 0; m0 < NPOS; m0 += 32) {
        // prefetch next K tile while this one is consumed
        if (m0 + 32 < NPOS) {
            __builtin_prefetch(kbm + (g * NPOS + m0 + 32 + (tid >> 3)) * DHEAD, 0, 0);
        }
        __syncthreads();
        {   // stage K (transposed) and V tiles: 32 rows x 64 d, 8 elems/thread
            int mrow = tid >> 3;
            int db   = (tid & 7) * 8;
            const unsigned short* ksrc = kbm + (g * NPOS + m0 + mrow) * DHEAD + db;
            const unsigned short* vsrc = vbm + (g * NPOS + m0 + mrow) * DHEAD + db;
            // V: fire-and-forget DMA into LDS (ASYNCcnt)
            async_copy_b128(&vt[mrow][db], vsrc);
            // K: transpose through VGPRs while the async copy is in flight
            union { uint4 u; unsigned short h[8]; } tk;
            tk.u = *(const uint4*)ksrc;
#pragma unroll
            for (int i = 0; i < 8; ++i) ktT[db + i][mrow] = tk.h[i];
            async_wait0();
        }
        __syncthreads();

        float kn_m0 = kn[g * NPOS + m0 + (lane & 15)];
        float kn_m1 = kn[g * NPOS + m0 + 16 + (lane & 15)];

#pragma unroll
        for (int ms = 0; ms < 2; ++ms) {
            AccF s;
#pragma unroll
            for (int r = 0; r < 8; ++r) s.f[r] = 0.0f;
#pragma unroll
            for (int dk = 0; dk < 2; ++dk) {
                FragBF b;   // lane&15 = m column; elements = d (table order)
#pragma unroll
                for (int i = 0; i < 16; ++i)
                    b.h[i] = ktT[dk * 32 + (lane >> 4) * 8 + (i & 7) + ((i >> 3) << 4)][ms * 16 + (lane & 15)];
                s.v = __builtin_amdgcn_wmma_f32_16x16x32_bf16(
                    false, qa[dk].v, false, b.v, (short)0, s.v, false, false);
            }
            // cosine scaling + stash P tile (bf16) in per-wave LDS
            float knv = (ms == 0) ? kn_m0 : kn_m1;
            int col = ms * 16 + (lane & 15);
            int rb  = (lane >> 4) * 8;
#pragma unroll
            for (int r = 0; r < 8; ++r) {
                float p = s.f[r] / (qn_r[r] * knv + SMOOTH);
                pst[wv][rb + r][col] = f2bf(p);
            }
        }

        // reload P in A-matrix swizzle (wave-private LDS, in-order DS pipe)
        FragBF pa;
        {
            int row = lane & 15;
            int kbo = (lane >> 4) * 8;
            pa.q[0] = *(const uint4*)&pst[wv][row][kbo];
            pa.q[1] = *(const uint4*)&pst[wv][row][kbo + 16];
        }
#pragma unroll
        for (int dn = 0; dn < 4; ++dn) {
            FragBF b;   // lane&15 = d column; elements = m (table order)
#pragma unroll
            for (int i = 0; i < 16; ++i)
                b.h[i] = vt[(lane >> 4) * 8 + (i & 7) + ((i >> 3) << 4)][dn * 16 + (lane & 15)];
            oacc[dn].v = __builtin_amdgcn_wmma_f32_16x16x32_bf16(
                false, pa.v, false, b.v, (short)0, oacc[dn].v, false, false);
        }
    }

    // write O tile channel-major: row = g*64+d, 8 consecutive positions/lane
    {
        int rb = (lane >> 4) * 8;
#pragma unroll
        for (int dn = 0; dn < 4; ++dn) {
            int d = dn * 16 + (lane & 15);
            union { uint4 u; unsigned short h[8]; } pk;
#pragma unroll
            for (int r = 0; r < 8; ++r) pk.h[r] = f2bf(oacc[dn].f[r]);
            *(uint4*)&ob[(g * DHEAD + d) * NPOS + n0w + rb] = pk.u;
        }
    }
}

// ---------------------------------------------------------------------------
// Kernel 4: projection  proj[c,p] = sum_i w_out[c,i] * O[i,p]
//           (M=256, K=512, N=4096), O is channel-major [512][4096] bf16.
// ---------------------------------------------------------------------------
__global__ __launch_bounds__(256) void k_proj(
    const unsigned short* __restrict__ ob,   // [512][4096] bf16
    const float* __restrict__ w,             // [256][512]
    float* __restrict__ proj)                // [256][4096]
{
    __shared__ __align__(16) unsigned short at[128][32];
    __shared__ __align__(16) unsigned short bt[32][64];

    const int tid  = threadIdx.x;
    const int lane = tid & 31;
    const int wv   = tid >> 5;
    const int mblk = blockIdx.x & 1;
    const int nblk = blockIdx.x >> 1;
    const int c0   = mblk * 128;
    const int p0   = nblk * 64;

    AccF acc[4];
#pragma unroll
    for (int i = 0; i < 4; ++i)
#pragma unroll
        for (int r = 0; r < 8; ++r) acc[i].f[r] = 0.0f;

    for (int k0 = 0; k0 < INNER; k0 += 32) {
        __syncthreads();
        {   // A tile from w_out (fp32 -> bf16)
            int row = tid >> 1;
            int cb  = (tid & 1) * 16;
            const float* src = w + (c0 + row) * INNER + k0 + cb;
#pragma unroll
            for (int i = 0; i < 16; ++i) at[row][cb + i] = f2bf(src[i]);
        }
        {   // B tile: contiguous rows of channel-major O, async DMA into LDS
            int kl = tid >> 3;
            int cb = (tid & 7) * 8;
            async_copy_b128(&bt[kl][cb], ob + (k0 + kl) * NPOS + p0 + cb);
            async_wait0();
        }
        __syncthreads();

        FragBF a;
        {
            int r   = wv * 16 + (lane & 15);
            int kbo = (lane >> 4) * 8;
            a.q[0] = *(const uint4*)&at[r][kbo];
            a.q[1] = *(const uint4*)&at[r][kbo + 16];
        }
#pragma unroll
        for (int ns = 0; ns < 4; ++ns) {
            FragBF b;
#pragma unroll
            for (int i = 0; i < 16; ++i)
                b.h[i] = bt[(lane >> 4) * 8 + (i & 7) + ((i >> 3) << 4)][ns * 16 + (lane & 15)];
            acc[ns].v = __builtin_amdgcn_wmma_f32_16x16x32_bf16(
                false, a.v, false, b.v, (short)0, acc[ns].v, false, false);
        }
    }

    {
        int rb = wv * 16 + ((lane >> 4) * 8);
#pragma unroll
        for (int ns = 0; ns < 4; ++ns) {
            int p = p0 + ns * 16 + (lane & 15);
#pragma unroll
            for (int r = 0; r < 8; ++r)
                proj[(c0 + rb + r) * NPOS + p] = acc[ns].f[r];
        }
    }
}

// ---------------------------------------------------------------------------
// Kernel 5: per-channel BatchNorm (training stats) + affine + ReLU.
// One workgroup per channel (256 channels x 4096 positions).
// ---------------------------------------------------------------------------
__global__ __launch_bounds__(256) void k_bn(
    const float* __restrict__ proj,
    const float* __restrict__ gamma,
    const float* __restrict__ beta,
    float* __restrict__ out)
{
    __shared__ float red1[256];
    __shared__ float red2[256];
    const int c   = blockIdx.x;
    const int tid = threadIdx.x;
    const float* row = proj + c * NPOS;

    float s = 0.0f, s2 = 0.0f;
    for (int i = tid; i < NPOS; i += 256) {
        float v = row[i];
        s += v;
        s2 += v * v;
    }
    red1[tid] = s;
    red2[tid] = s2;
    __syncthreads();
    for (int off = 128; off > 0; off >>= 1) {
        if (tid < off) {
            red1[tid] += red1[tid + off];
            red2[tid] += red2[tid + off];
        }
        __syncthreads();
    }
    float mean = red1[0] * (1.0f / NPOS);
    float var  = red2[0] * (1.0f / NPOS) - mean * mean;
    float inv  = rsqrtf(var + BN_EPS);
    float ga   = gamma[c];
    float be   = beta[c];
    for (int i = tid; i < NPOS; i += 256) {
        float v = (row[i] - mean) * inv * ga + be;
        out[c * NPOS + i] = v > 0.0f ? v : 0.0f;
    }
}

// ---------------------------------------------------------------------------
// Host-side launch. Inputs: x, w_qkv, w_out, gamma, beta (all fp32).
// Workspace layout (bytes):
//   [0,4M)   q bf16   [4M,8M)  k bf16   [8M,12M) v bf16   [12M,16M) O bf16
//   [16M, +128K) qn   [+128K, +256K) kn   [+256K, +4M+256K) proj fp32
// ---------------------------------------------------------------------------
extern "C" void kernel_launch(void* const* d_in, const int* in_sizes, int n_in,
                              void* d_out, int out_size, void* d_ws, size_t ws_size,
                              hipStream_t stream) {
    (void)in_sizes; (void)n_in; (void)out_size; (void)ws_size;
    const float* x      = (const float*)d_in[0];
    const float* w_qkv  = (const float*)d_in[1];
    const float* w_out  = (const float*)d_in[2];
    const float* gamma  = (const float*)d_in[3];
    const float* beta   = (const float*)d_in[4];
    float* out = (float*)d_out;

    char* ws = (char*)d_ws;
    unsigned short* qb = (unsigned short*)(ws + (size_t)0);
    unsigned short* kb = (unsigned short*)(ws + ((size_t)4 << 20));
    unsigned short* vb = (unsigned short*)(ws + ((size_t)8 << 20));
    unsigned short* ob = (unsigned short*)(ws + ((size_t)12 << 20));
    float* qn   = (float*)(ws + ((size_t)16 << 20));
    float* kn   = (float*)(ws + ((size_t)16 << 20) + ((size_t)128 << 10));
    float* proj = (float*)(ws + ((size_t)16 << 20) + ((size_t)256 << 10));

    k_qkv_gemm<<<768, 256, 0, stream>>>(x, w_qkv, qb, kb, vb);
    k_norms  <<<128, 256, 0, stream>>>(qb, kb, qn, kn);
    k_attn   <<<256, 256, 0, stream>>>(qb, kb, vb, qn, kn, ob);
    k_proj   <<<128, 256, 0, stream>>>(ob, w_out, proj);
    k_bn     <<<256, 256, 0, stream>>>(proj, gamma, beta, out);
}
